// DSCBlockLBP_3745211482159
// MI455X (gfx1250) — matrix operand order repaired and checked by reading, not compiled, not run on hardware
//
#include <hip/hip_runtime.h>

typedef __attribute__((ext_vector_type(16))) __bf16 v16bf;
typedef __attribute__((ext_vector_type(8)))  __bf16 v8bf;
typedef __attribute__((ext_vector_type(8)))  float  v8f;

#define BATCH 8
#define CH    256
#define NWW   512
#define HH    128
#define WW    128
#define HWSZ  (HH*WW)
#define EPSV  1e-5f

// spatial tile per block: 8 rows x 16 cols
#define TH 8
#define TW 16
#define XS_RS 18                 // (TW+2) floats per halo row
#define XS_CHN (10*XS_RS)        // 180 floats per channel plane
#define YT_RS 520                // NWW + 8 pad (16B-aligned rows, spreads banks)
#define XS_BYTES (CH*XS_CHN*4)        // 184320
#define YT_BYTES (TH*TW*YT_RS*2)      // 133120
#define SMEM_BYTES (XS_BYTES + YT_BYTES)  // 317440 <= 320KB

static __device__ __forceinline__ unsigned short f32_to_bf16(float f) {
    unsigned int u = __float_as_uint(f);
    unsigned int r = (u + 0x7FFFu + ((u >> 16) & 1u)) >> 16;   // round-nearest-even
    return (unsigned short)r;
}

// ---------------------------------------------------------------------------
// Kernel 1: per-channel BN stats -> scale[c] = gamma*rsqrt(var+eps),
//           shift[c] = beta - mean*scale   (biased var, training mode)
// float4 loads keep this memory-bound instead of issue-bound.
// ---------------------------------------------------------------------------
__global__ void __launch_bounds__(256)
bn_stats_kernel(const float* __restrict__ x, const float* __restrict__ gamma,
                const float* __restrict__ beta, float* __restrict__ scale,
                float* __restrict__ shift) {
    __shared__ float s_sum[256];
    __shared__ float s_sq[256];
    const int c = blockIdx.x;
    const int t = threadIdx.x;
    float sum = 0.f, sq = 0.f;
    const float4* x4 = (const float4*)x;
    for (int k = t; k < (BATCH * HWSZ) / 4; k += 256) {
        int b   = k >> 12;                 // HWSZ/4 = 4096 per (b,c)
        int hw4 = k & 4095;
        float4 v = x4[(((size_t)b * CH + c) << 12) + hw4];
        sum += v.x + v.y + v.z + v.w;
        sq  += v.x * v.x + v.y * v.y + v.z * v.z + v.w * v.w;
    }
    s_sum[t] = sum; s_sq[t] = sq;
    __syncthreads();
    for (int off = 128; off > 0; off >>= 1) {
        if (t < off) { s_sum[t] += s_sum[t + off]; s_sq[t] += s_sq[t + off]; }
        __syncthreads();
    }
    if (t == 0) {
        const float inv_n = 1.0f / (float)(BATCH * HWSZ);
        float mean = s_sum[0] * inv_n;
        float var  = s_sq[0] * inv_n - mean * mean;
        float inv  = rsqrtf(var + EPSV);
        float sc   = gamma[c] * inv;
        scale[c] = sc;
        shift[c] = beta[c] - mean * sc;
    }
}

// ---------------------------------------------------------------------------
// Kernel 2: shuffle w1 (C x NW f32) into bf16 WMMA A-fragment order
// (16-bit A 16x32 layout, CDNA5 ISA 7.12.2) -> each lane loads 32B contig.
// ---------------------------------------------------------------------------
__global__ void __launch_bounds__(256)
prep_w1_kernel(const float* __restrict__ w1, unsigned short* __restrict__ w1p) {
    int o    = blockIdx.x * 256 + threadIdx.x;      // 131072 total
    int i    = o & 15;
    int lane = (o >> 4) & 31;
    int ks   = (o >> 9) & 15;
    int mt   = o >> 13;
    int c    = mt * 16 + (lane & 15);
    int half = lane >> 4;
    int nw   = ks * 32 + ((i >> 3) << 4) + (i & 7) + half * 8;
    w1p[o] = f32_to_bf16(w1[c * NWW + nw]);
}

// ---------------------------------------------------------------------------
// Kernel 3: fused  BN-normalize + depthwise 3x3 (ternary) + ReLU + 1x1 GEMM
//           (bf16 WMMA, f32 accum) + bias + residual.
// One block per (batch, 8x16 spatial tile); 512 threads = 16 waves.
// ---------------------------------------------------------------------------
__global__ void __launch_bounds__(512)
fused_dsc_kernel(const float* __restrict__ x, const float* __restrict__ lbp,
                 const float* __restrict__ b1, const float* __restrict__ scale,
                 const float* __restrict__ shift,
                 const unsigned short* __restrict__ w1p,
                 float* __restrict__ out) {
    extern __shared__ char smem[];
    float*          xs = (float*)smem;                        // raw x tile + halo
    unsigned short* yt = (unsigned short*)(smem + XS_BYTES);  // bf16 y, [p][nw]

    const int bid  = blockIdx.x;
    const int wt   = bid & 7;           // W/16 = 8 tiles
    const int htl  = (bid >> 3) & 15;   // H/8  = 16 tiles
    const int b    = bid >> 7;
    const int h0   = htl * TH;
    const int w0   = wt * TW;
    const int tid  = threadIdx.x;
    const int lane = tid & 31;
    const int wave = tid >> 5;

    // ---- phase 1: stage raw x tile (+1 halo, zero-padded), one row/thread ----
    {
        const float* xb = x + ((size_t)b * CH << 14);
        for (int task = tid; task < CH * 10; task += 512) {   // 2560/512 = 5, uniform
            int c  = task / 10;
            int r  = task - c * 10;
            int gh = h0 - 1 + r;
            float vals[18];
            if ((unsigned)gh < (unsigned)HH) {
                const float* row = xb + ((size_t)c << 14) + gh * WW;
                const float4* rm = (const float4*)(row + w0);   // 64B aligned
                float4 m0 = rm[0], m1 = rm[1], m2 = rm[2], m3 = rm[3];
                vals[0]  = (w0 > 0) ? row[w0 - 1] : 0.f;
                vals[17] = (w0 + 16 < WW) ? row[w0 + 16] : 0.f;
                vals[1] = m0.x;  vals[2] = m0.y;  vals[3] = m0.z;  vals[4] = m0.w;
                vals[5] = m1.x;  vals[6] = m1.y;  vals[7] = m1.z;  vals[8] = m1.w;
                vals[9] = m2.x;  vals[10] = m2.y; vals[11] = m2.z; vals[12] = m2.w;
                vals[13] = m3.x; vals[14] = m3.y; vals[15] = m3.z; vals[16] = m3.w;
            } else {
                #pragma unroll
                for (int j = 0; j < 18; ++j) vals[j] = 0.f;
            }
            float* dst = xs + c * XS_CHN + r * XS_RS;          // 8B aligned rows
            #pragma unroll
            for (int j = 0; j < 18; j += 2)
                *(float2*)(dst + j) = make_float2(vals[j], vals[j + 1]);
        }
    }
    __syncthreads();

    // ---- phase 2: depthwise 3x3 + BN affine + ReLU -> yt (transposed bf16) ----
    // sliding 3x18 register window: each halo row loaded once, reused 3x16 times
    {
        const int nw = tid;              // one depthwise channel per thread
        const int c  = nw >> 1;
        float wgt[9];
        float wsum = 0.f;
        #pragma unroll
        for (int j = 0; j < 9; ++j) { wgt[j] = lbp[nw * 9 + j]; wsum += wgt[j]; }
        const float sc  = scale[c];
        const float shw = shift[c] * wsum;   // BN folded into the conv
        const float* xc = xs + c * XS_CHN;

        float win[3][18];
        #pragma unroll
        for (int j = 0; j < 18; ++j) win[0][j] = xc[j];
        #pragma unroll
        for (int j = 0; j < 18; ++j) win[1][j] = xc[XS_RS + j];

        #pragma unroll
        for (int ph = 0; ph < TH; ++ph) {
            float* rN = win[(ph + 2) % 3];
            #pragma unroll
            for (int j = 0; j < 18; ++j) rN[j] = xc[(ph + 2) * XS_RS + j];
            const float* r0 = win[ph % 3];
            const float* r1 = win[(ph + 1) % 3];
            const float* r2 = win[(ph + 2) % 3];
            #pragma unroll
            for (int pw = 0; pw < TW; ++pw) {
                float acc;
                acc = wgt[0] * r0[pw];
                acc = fmaf(wgt[1], r0[pw + 1], acc);
                acc = fmaf(wgt[2], r0[pw + 2], acc);
                acc = fmaf(wgt[3], r1[pw],     acc);
                acc = fmaf(wgt[4], r1[pw + 1], acc);
                acc = fmaf(wgt[5], r1[pw + 2], acc);
                acc = fmaf(wgt[6], r2[pw],     acc);
                acc = fmaf(wgt[7], r2[pw + 1], acc);
                acc = fmaf(wgt[8], r2[pw + 2], acc);
                float y = fmaf(sc, acc, shw);
                y = fmaxf(y, 0.f);
                yt[(ph * TW + pw) * YT_RS + nw] = f32_to_bf16(y);
            }
        }
    }
    __syncthreads();

    // ---- phase 3: GEMM  z[c,p] = sum_nw w1[c,nw] * y[nw,p]  via bf16 WMMA ----
    const int mt   = wave;          // c-tile (16 rows), 16 waves cover C=256
    const int half = lane >> 4;
    const int lcol = lane & 15;

    v8f acc[8];
    #pragma unroll
    for (int ns = 0; ns < 8; ++ns)
        #pragma unroll
        for (int r = 0; r < 8; ++r) acc[ns][r] = 0.f;

    const v8bf* __restrict__ w1v = (const v8bf*)w1p;
    for (int ks = 0; ks < 16; ++ks) {
        // A fragment: 32 contiguous bytes per lane from pre-shuffled w1 (L2)
        int aidx = ((mt * 16 + ks) * 32 + lane) * 2;
        v8bf alo = w1v[aidx];
        v8bf ahi = w1v[aidx + 1];
        v16bf a = __builtin_shufflevector(alo, ahi,
                    0,1,2,3,4,5,6,7,8,9,10,11,12,13,14,15);
        #pragma unroll
        for (int ns = 0; ns < 8; ++ns) {
            // B fragment: lane = column p, K (nw) = 8*half+[0..7] and 16+8*half+[0..7], LDS
            const unsigned short* bp =
                yt + (ns * 16 + lcol) * YT_RS + ks * 32 + half * 8;
            v8bf blo = *(const v8bf*)bp;
            v8bf bhi = *(const v8bf*)(bp + 16);
            v16bf bb = __builtin_shufflevector(blo, bhi,
                        0,1,2,3,4,5,6,7,8,9,10,11,12,13,14,15);
            acc[ns] = __builtin_amdgcn_wmma_f32_16x16x32_bf16(
                          false, a, false, bb, (short)0, acc[ns], false, false);
        }
    }

    // ---- epilogue: + b1 + residual (raw x from LDS), store z ----
    float bias[8];
    #pragma unroll
    for (int r = 0; r < 8; ++r) bias[r] = b1[mt * 16 + half * 8 + r];

    #pragma unroll
    for (int ns = 0; ns < 8; ++ns) {      // ns == tile row
        #pragma unroll
        for (int r = 0; r < 8; ++r) {
            int c = mt * 16 + half * 8 + r;             // C/D layout: M = r + 8*(lane>>4)
            float resid = xs[c * XS_CHN + (ns + 1) * XS_RS + (lcol + 1)];
            float v = acc[ns][r] + bias[r] + resid;
            out[(((size_t)b * CH + c) << 14) + (h0 + ns) * WW + (w0 + lcol)] = v;
        }
    }
}

// ---------------------------------------------------------------------------
extern "C" void kernel_launch(void* const* d_in, const int* in_sizes, int n_in,
                              void* d_out, int out_size, void* d_ws, size_t ws_size,
                              hipStream_t stream) {
    const float* x     = (const float*)d_in[0];
    const float* gamma = (const float*)d_in[1];
    const float* beta  = (const float*)d_in[2];
    const float* lbp   = (const float*)d_in[3];
    const float* w1    = (const float*)d_in[4];
    const float* b1    = (const float*)d_in[5];
    float* out = (float*)d_out;

    float*          scale = (float*)d_ws;
    float*          shift = (float*)((char*)d_ws + 1024);
    unsigned short* w1p   = (unsigned short*)((char*)d_ws + 2048);  // 256 KB

    bn_stats_kernel<<<CH, 256, 0, stream>>>(x, gamma, beta, scale, shift);
    prep_w1_kernel<<<(CH * NWW) / 256, 256, 0, stream>>>(w1, w1p);

    (void)hipFuncSetAttribute((const void*)fused_dsc_kernel,
                              hipFuncAttributeMaxDynamicSharedMemorySize,
                              SMEM_BYTES);
    const int nblocks = BATCH * (HH / TH) * (WW / TW);   // 1024
    fused_dsc_kernel<<<nblocks, 512, SMEM_BYTES, stream>>>(
        x, lbp, b1, scale, shift, w1p, out);
}
